// AdditiveAttention_61641370632425
// MI455X (gfx1250) — matrix operand order repaired and checked by reading, not compiled, not run on hardware
//
#include <hip/hip_runtime.h>

#define B_ 64
#define S_ 2048
#define H_ 512
#define E_ 1024  // 2*H

typedef __attribute__((ext_vector_type(16))) __bf16          v16bf;
typedef __attribute__((ext_vector_type(8)))  float           v8f;
typedef __attribute__((ext_vector_type(8)))  unsigned short  v8us;
typedef __attribute__((ext_vector_type(16))) unsigned short  v16us;

// ---------------- kernel 0a: proj_dec = dec @ W_d  (B x H) ----------------
__global__ __launch_bounds__(256)
void k_projdec(const float* __restrict__ dec, const float* __restrict__ Wd,
               float* __restrict__ pdec) {
    int b = blockIdx.x, tid = threadIdx.x;
    __shared__ float sdec[H_];
    for (int i = tid; i < H_; i += 256) sdec[i] = dec[b * H_ + i];
    __syncthreads();
    for (int h = tid; h < H_; h += 256) {
        float acc = 0.f;
        #pragma unroll 8
        for (int d = 0; d < H_; ++d) acc = fmaf(sdec[d], Wd[(size_t)d * H_ + h], acc);
        pdec[b * H_ + h] = acc;
    }
}

// ------ kernel 0b: swizzle W_h (2H x H, f32 row-major) into bf16 WMMA-B layout ------
// B operand of v_wmma_f32_16x16x32_bf16: lane holds column n = lane%16;
// lanes 0-15 hold K {0..7,16..23}, lanes 16-31 hold K {8..15,24..31}; VGPR r = pair (2r, 2r+1).
// Stored per (ktile, ntile) as 32 lanes x 16 bf16 contiguous.
__global__ __launch_bounds__(32)
void k_swizzle(const float* __restrict__ Wh, __bf16* __restrict__ Wb) {
    int kt = blockIdx.x;              // 0..31, K tiles of 32
    int nt = blockIdx.y;              // 0..31, N tiles of 16
    int lane = threadIdx.x;
    int n  = nt * 16 + (lane & 15);
    int kb = kt * 32 + ((lane < 16) ? 0 : 8);
    __bf16* dst = Wb + (((size_t)kt * 32 + nt) * 32 + lane) * 16;
    #pragma unroll
    for (int r = 0; r < 8; ++r) {
        int k = kb + 2 * r + ((r >= 4) ? 8 : 0);
        dst[2 * r    ] = (__bf16)Wh[(size_t)k       * H_ + n];
        dst[2 * r + 1] = (__bf16)Wh[(size_t)(k + 1) * H_ + n];
    }
}

// ---------------- kernel 1: fused GEMM + tanh + v-dot + mask -> scores ----------------
// Workgroup = (batch b, 32-row S tile). 8 waves; wave w owns columns [w*64, w*64+64)
// and both 16-row M-tiles (each B tile feeds 2 WMMAs -> half the L2 B traffic).
__global__ __launch_bounds__(256)
void k_score(const float* __restrict__ enc, const float* __restrict__ pdec,
             const __bf16* __restrict__ Wb, const float* __restrict__ vvec,
             const int* __restrict__ mask, float* __restrict__ scores) {
    const int b    = blockIdx.x;
    const int m0   = blockIdx.y * 32;
    const int tid  = threadIdx.x;
    const int wave = tid >> 5;
    const int lane = tid & 31;

    __shared__ __align__(16) __bf16 As[32 * 1032];   // 32 rows x 1024 K, pad 8
    __shared__ float s_part[8][32];

    // Stage A-panel: 32 contiguous rows x 1024 f32 -> bf16 LDS (coalesced float4)
    const float4* src = (const float4*)(enc + ((size_t)b * S_ + m0) * E_);
    #pragma unroll
    for (int it = 0; it < 32; ++it) {
        int i4 = it * 256 + tid;              // 0..8191
        float4 f = src[i4];
        int r = i4 >> 8;                      // row 0..31 (256 float4 per row)
        int c = (i4 & 255) * 4;
        __bf16* d = &As[r * 1032 + c];
        d[0] = (__bf16)f.x; d[1] = (__bf16)f.y;
        d[2] = (__bf16)f.z; d[3] = (__bf16)f.w;
    }
    __syncthreads();

    // ISA 16-bit A layout: lane m=lane%16; lanes<16 hold K {0..7,16..23},
    // lanes>=16 hold K {8..15,24..31}; two 16B chunks per K-step.
    const int m = lane & 15;
    const __bf16* abase0 = As + m * 1032 + ((lane < 16) ? 0 : 8);   // rows 0..15
    const __bf16* abase1 = abase0 + 16 * 1032;                      // rows 16..31
    const __bf16* bbase  = Wb + ((size_t)(wave * 4) * (32 * 16) + (size_t)lane * 16);

    v8f acc[2][4];
    #pragma unroll
    for (int mt = 0; mt < 2; ++mt)
        #pragma unroll
        for (int j = 0; j < 4; ++j) acc[mt][j] = (v8f){};

    for (int kk = 0; kk < 32; ++kk) {
        v8us a0lo = *(const v8us*)(abase0 + kk * 32);
        v8us a0hi = *(const v8us*)(abase0 + kk * 32 + 16);
        v8us a1lo = *(const v8us*)(abase1 + kk * 32);
        v8us a1hi = *(const v8us*)(abase1 + kk * 32 + 16);
        v16bf a0 = __builtin_bit_cast(v16bf, __builtin_shufflevector(a0lo, a0hi,
                     0,1,2,3,4,5,6,7,8,9,10,11,12,13,14,15));
        v16bf a1 = __builtin_bit_cast(v16bf, __builtin_shufflevector(a1lo, a1hi,
                     0,1,2,3,4,5,6,7,8,9,10,11,12,13,14,15));

        const __bf16* bp = bbase + (size_t)kk * (32 * 32 * 16);
        v16bf b0 = *(const v16bf*)(bp);
        v16bf b1 = *(const v16bf*)(bp + 1 * 32 * 16);
        v16bf b2 = *(const v16bf*)(bp + 2 * 32 * 16);
        v16bf b3 = *(const v16bf*)(bp + 3 * 32 * 16);

        acc[0][0] = __builtin_amdgcn_wmma_f32_16x16x32_bf16(false, a0, false, b0, (short)0, acc[0][0], false, false);
        acc[1][0] = __builtin_amdgcn_wmma_f32_16x16x32_bf16(false, a1, false, b0, (short)0, acc[1][0], false, false);
        acc[0][1] = __builtin_amdgcn_wmma_f32_16x16x32_bf16(false, a0, false, b1, (short)0, acc[0][1], false, false);
        acc[1][1] = __builtin_amdgcn_wmma_f32_16x16x32_bf16(false, a1, false, b1, (short)0, acc[1][1], false, false);
        acc[0][2] = __builtin_amdgcn_wmma_f32_16x16x32_bf16(false, a0, false, b2, (short)0, acc[0][2], false, false);
        acc[1][2] = __builtin_amdgcn_wmma_f32_16x16x32_bf16(false, a1, false, b2, (short)0, acc[1][2], false, false);
        acc[0][3] = __builtin_amdgcn_wmma_f32_16x16x32_bf16(false, a0, false, b3, (short)0, acc[0][3], false, false);
        acc[1][3] = __builtin_amdgcn_wmma_f32_16x16x32_bf16(false, a1, false, b3, (short)0, acc[1][3], false, false);
    }

    // Epilogue: C layout -> lane holds column n, VGPR r is row (r + (lane<16?0:8)).
    float rowsum[2][8];
    #pragma unroll
    for (int mt = 0; mt < 2; ++mt)
        #pragma unroll
        for (int r = 0; r < 8; ++r) rowsum[mt][r] = 0.f;

    #pragma unroll
    for (int j = 0; j < 4; ++j) {
        int n = wave * 64 + j * 16 + m;
        float pd = pdec[b * H_ + n];
        float vn = vvec[n];
        #pragma unroll
        for (int mt = 0; mt < 2; ++mt)
            #pragma unroll
            for (int r = 0; r < 8; ++r)
                rowsum[mt][r] += tanhf(acc[mt][j][r] + pd) * vn;
    }

    // Butterfly reduce over the 16 lanes of each half-wave (halves stay disjoint)
    #pragma unroll
    for (int mt = 0; mt < 2; ++mt)
        #pragma unroll
        for (int r = 0; r < 8; ++r) {
            float t = rowsum[mt][r];
            t += __shfl_xor(t, 1);
            t += __shfl_xor(t, 2);
            t += __shfl_xor(t, 4);
            t += __shfl_xor(t, 8);
            rowsum[mt][r] = t;
        }
    if (m == 0) {
        int rb = (lane < 16) ? 0 : 8;
        #pragma unroll
        for (int mt = 0; mt < 2; ++mt)
            #pragma unroll
            for (int r = 0; r < 8; ++r)
                s_part[wave][mt * 16 + rb + r] = rowsum[mt][r];
    }
    __syncthreads();

    if (tid < 32) {
        float s = 0.f;
        #pragma unroll
        for (int w = 0; w < 8; ++w) s += s_part[w][tid];
        int mk = mask[b * S_ + m0 + tid];
        scores[b * S_ + m0 + tid] = (mk == 0) ? -10000.0f : s;
    }
}

// ---------------- kernel 2: masked softmax over S per batch ----------------
__global__ __launch_bounds__(256)
void k_softmax(const float* __restrict__ scores, float* __restrict__ attn) {
    int b = blockIdx.x, tid = threadIdx.x;
    __shared__ float sc[S_];
    __shared__ float red[8];

    float lmax = -3.0e38f;
    for (int i = tid; i < S_; i += 256) {
        float s = scores[b * S_ + i];
        sc[i] = s;
        lmax = fmaxf(lmax, s);
    }
    #pragma unroll
    for (int msk = 16; msk >= 1; msk >>= 1) lmax = fmaxf(lmax, __shfl_xor(lmax, msk));
    if ((tid & 31) == 0) red[tid >> 5] = lmax;
    __syncthreads();
    float gmax = red[0];
    #pragma unroll
    for (int w = 1; w < 8; ++w) gmax = fmaxf(gmax, red[w]);

    float lsum = 0.f;
    for (int i = tid; i < S_; i += 256) {
        float e = __expf(sc[i] - gmax);
        sc[i] = e;
        lsum += e;
    }
    #pragma unroll
    for (int msk = 16; msk >= 1; msk >>= 1) lsum += __shfl_xor(lsum, msk);
    __syncthreads();
    if ((tid & 31) == 0) red[tid >> 5] = lsum;
    __syncthreads();
    float gsum = 0.f;
    #pragma unroll
    for (int w = 0; w < 8; ++w) gsum += red[w];
    float inv = 1.0f / gsum;

    for (int i = tid; i < S_; i += 256) attn[b * S_ + i] = sc[i] * inv;
}

// ---------------- kernel 3: context = attn @ enc  (B x E) ----------------
__global__ __launch_bounds__(256)
void k_context(const float* __restrict__ enc, const float* __restrict__ attn,
               float* __restrict__ ctx) {
    int b = blockIdx.x;
    int e = blockIdx.y * 256 + threadIdx.x;
    __shared__ float w[S_];
    for (int i = threadIdx.x; i < S_; i += 256) w[i] = attn[b * S_ + i];
    __syncthreads();
    const float* ep = enc + (size_t)b * S_ * E_ + e;
    float acc = 0.f;
    #pragma unroll 8
    for (int s = 0; s < S_; ++s) acc = fmaf(w[s], ep[(size_t)s * E_], acc);
    ctx[b * E_ + e] = acc;
}

// ---------------- host launcher ----------------
extern "C" void kernel_launch(void* const* d_in, const int* in_sizes, int n_in,
                              void* d_out, int out_size, void* d_ws, size_t ws_size,
                              hipStream_t stream) {
    const float* dec  = (const float*)d_in[0];   // (B, H)
    const float* enc  = (const float*)d_in[1];   // (B, S, 2H)
    const int*   mask = (const int*)  d_in[2];   // (B, S)
    const float* Wh   = (const float*)d_in[3];   // (2H, H)
    const float* Wd   = (const float*)d_in[4];   // (H, H)
    const float* v    = (const float*)d_in[5];   // (H,)

    float*  scores = (float*)d_ws;                       // B*S f32
    float*  pdec   = scores + (size_t)B_ * S_;           // B*H f32
    __bf16* Wb     = (__bf16*)(pdec + (size_t)B_ * H_);  // 2H*H bf16 (swizzled)

    float* out  = (float*)d_out;
    float* ctx  = out;                    // B*E floats (context first)
    float* attn = out + (size_t)B_ * E_;  // B*S floats (attn_weights second)

    k_projdec<<<B_, 256, 0, stream>>>(dec, Wd, pdec);
    k_swizzle<<<dim3(32, 32), 32, 0, stream>>>(Wh, Wb);
    k_score<<<dim3(B_, S_ / 32), 256, 0, stream>>>(enc, pdec, Wb, v, mask, scores);
    k_softmax<<<B_, 256, 0, stream>>>(scores, attn);
    k_context<<<dim3(B_, E_ / 256), 256, 0, stream>>>(enc, attn, ctx);
}
